// RouteNet_Fermi_56985626083689
// MI455X (gfx1250) — matrix operand order repaired and checked by reading, not compiled, not run on hardware
//
#include <hip/hip_runtime.h>
#include <hip/hip_bf16.h>

#define PN 50000
#define TN 8
#define QN 20000
#define LN 8000
#define KQn 32
#define KLn 32
#define Dn 32
#define ITERS 8

typedef __attribute__((ext_vector_type(16))) __bf16 v16bf;
typedef __attribute__((ext_vector_type(8)))  __bf16 v8bf;
typedef __attribute__((ext_vector_type(8)))  float  v8f;
typedef __attribute__((ext_vector_type(4)))  unsigned int v4u;
typedef __attribute__((ext_vector_type(8)))  int v8i;
typedef __attribute__((ext_vector_type(4)))  int v4i;

__device__ __forceinline__ __bf16 f2bf(float f) {
  unsigned u = __float_as_uint(f);
  unsigned short s = (unsigned short)((u + 0x7fffu + ((u >> 16) & 1u)) >> 16);  // RNE
  return __builtin_bit_cast(__bf16, s);
}

__device__ __forceinline__ float sigmoidf_(float x) {
  return __builtin_amdgcn_rcpf(1.0f + __expf(-x));
}
__device__ __forceinline__ float tanhf_(float x) {
  return 1.0f - 2.0f * __builtin_amdgcn_rcpf(1.0f + __expf(2.0f * x));
}

__device__ __forceinline__ v8f wmma_bf16(v16bf a, v16bf b, v8f c) {
  return __builtin_amdgcn_wmma_f32_16x16x32_bf16(false, a, false, b, (short)0, c, false, false);
}

// A operand from a row-major bf16 LDS tile [16][ldK]: per lane two contiguous
// 16B chunks of row (lane&15) -> 2x ds_load_b128, zero VALU.
__device__ __forceinline__ v16bf load_a_bf(const __bf16* tile, int ldK, int lane, int kslice) {
  int m = lane & 15;
  int kb = kslice * 32 + ((lane >> 4) << 3);
  const __bf16* p = tile + m * ldK + kb;
  union { v16bf v; v8bf h[2]; } u;
  u.h[0] = *(const v8bf*)(p);
  u.h[1] = *(const v8bf*)(p + 16);
  return u.v;
}

// Pre-packed B operand [kslice][ntile][lane(32)][16] bf16 -> one 32B load per lane.
__device__ __forceinline__ v16bf load_b(const unsigned short* pack, int ks, int nt, int lane) {
  return *(const v16bf*)(pack + ((((ks * 6 + nt) * 32) + lane) << 4));
}

template <int KS>
__device__ __forceinline__ void gru_accum(const __bf16* const* xs, const __bf16* htb,
                                          const unsigned short* sbIH,
                                          const unsigned short* sbHH, int lane,
                                          v8f acc_i[6], v8f acc_h[6]) {
  v16bf ah = load_a_bf(htb, Dn, lane, 0);
  v16bf ax[KS];
#pragma unroll
  for (int s = 0; s < KS; ++s) ax[s] = load_a_bf(xs[s], Dn, lane, 0);
#pragma unroll
  for (int g = 0; g < 6; ++g) {
#pragma unroll
    for (int s = 0; s < KS; ++s)
      acc_i[g] = wmma_bf16(ax[s], load_b(sbIH, s, g, lane), acc_i[g]);
    acc_h[g] = wmma_bf16(ah, load_b(sbHH, 0, g, lane), acc_h[g]);
  }
}

// Gate math on the f32 C/D layout: VGPR r, lane l -> M = r + 8*(l>>4), N = (l&15) + 16*tile.
template <bool LDSOUT>
__device__ __forceinline__ void gru_gates(const v8f* acc_i, const v8f* acc_h, const float brA[2],
                                          const float bzA[2], const float binA[2],
                                          const float bhnA[2], const float (*hold)[Dn],
                                          float* stateF, __bf16* stateBF, __bf16 (*htb)[Dn],
                                          int row0, int lane) {
  int c0 = lane & 15, hi = lane >> 4;
#pragma unroll
  for (int half = 0; half < 2; ++half) {
    int c = c0 + 16 * half;
#pragma unroll
    for (int r = 0; r < 8; ++r) {
      int m = r + (hi << 3);
      float rg = sigmoidf_(acc_i[half][r] + acc_h[half][r] + brA[half]);
      float zg = sigmoidf_(acc_i[2 + half][r] + acc_h[2 + half][r] + bzA[half]);
      float ng = tanhf_(acc_i[4 + half][r] + binA[half] + rg * (acc_h[4 + half][r] + bhnA[half]));
      float hv = ng + zg * (hold[m][c] - ng);
      __bf16 hb16 = f2bf(hv);
      stateF[(size_t)(row0 + m) * Dn + c] = hv;
      stateBF[(size_t)(row0 + m) * Dn + c] = hb16;
      if (LDSOUT) htb[m][c] = hb16;
    }
  }
}

// ---------------- Tensor Data Mover helpers ----------------
__device__ __forceinline__ unsigned lds_off(const void* p) {
  return (unsigned)(unsigned long long)p;  // low 32 bits of flat shared addr = LDS offset
}

__device__ __forceinline__ void tdm_issue(v4u g0, v8i g1, v4i g2, v4i g3) {
#if __clang_major__ >= 23
  v8i g4 = {0, 0, 0, 0, 0, 0, 0, 0};
  __builtin_amdgcn_tensor_load_to_lds(g0, g1, g2, g3, g4, 0);
#else
  __builtin_amdgcn_tensor_load_to_lds(g0, g1, g2, g3, 0);
#endif
}

// Gather-mode TDM: 16 row indices (16-bit), row width = rowU8 * 8 bytes, dest LDS.
__device__ __forceinline__ void tdm_gather16(const void* gbase, unsigned ldsoff,
                                             const int idx[16], int rowU8,
                                             unsigned tensorRows) {
  unsigned long long ga = (unsigned long long)gbase;
  v4u g0 = {1u | (1u << 31),  // count=1, gather_mode=1, 16-bit idx
            ldsoff, (unsigned)ga, (unsigned)(ga >> 32) | (2u << 30)};
  v8i g1 = {(int)(3u << 16),                      // data_size = 8B
            (int)((unsigned)rowU8 << 16),         // tensor_dim0[15:0] (bits 63:48)
            (int)((tensorRows & 0xffffu) << 16),  // dim0_hi=0 | tensor_dim1 lo
            (int)(((tensorRows >> 16) & 0xffffu) | ((unsigned)rowU8 << 16)),  // dim1 hi | tile_dim0
            16,                                   // tile_dim1 = #valid indices
            rowU8, 0, 0};                         // tensor_dim0_stride
  unsigned w[8];
#pragma unroll
  for (int i = 0; i < 8; ++i)
    w[i] = ((unsigned)idx[2 * i] & 0xffffu) | (((unsigned)idx[2 * i + 1] & 0xffffu) << 16);
  v4i g2 = {(int)w[0], (int)w[1], (int)w[2], (int)w[3]};
  v4i g3 = {(int)w[4], (int)w[5], (int)w[6], (int)w[7]};
  tdm_issue(g0, g1, g2, g3);
}

// Linear 1-D TDM load of units8*8 contiguous bytes into LDS.
__device__ __forceinline__ void tdm_linear(const void* gbase, unsigned ldsoff, int units8) {
  unsigned long long ga = (unsigned long long)gbase;
  v4u g0 = {1u, ldsoff, (unsigned)ga, (unsigned)(ga >> 32) | (2u << 30)};
  v8i g1 = {(int)(3u << 16),
            (int)(((unsigned)units8 & 0xffffu) << 16),
            (int)((((unsigned)units8 >> 16) & 0xffffu) | (1u << 16)),  // dim0 hi | tensor_dim1=1
            (int)(((unsigned)units8 & 0xffffu) << 16),                 // tile_dim0
            0,                                                          // tile_dim1 unused (1-D)
            units8, 0, 0};
  v4i z = {0, 0, 0, 0};
  tdm_issue(g0, g1, z, z);
}

struct TileBuf {
  __bf16 xq[16][Dn];
  __bf16 xl[16][Dn];
  __bf16 hb[16][Dn];
  float  hof[16][Dn];
};

__device__ __forceinline__ void stage_path_tile(int iter, int row0, TileBuf* b,
                                                const float* path_state, const __bf16* ps_bf,
                                                const float* path_emb, const __bf16* pemb_bf,
                                                const __bf16* qs_bf, const __bf16* ls_bf,
                                                const int* q2p, const int* l2p) {
  int qi[16], li[16];
#pragma unroll
  for (int r = 0; r < 16; ++r) { qi[r] = q2p[row0 + r]; li[r] = l2p[row0 + r]; }
  tdm_gather16(qs_bf, lds_off(&b->xq[0][0]), qi, 8, QN);
  tdm_gather16(ls_bf, lds_off(&b->xl[0][0]), li, 8, LN);
  if (iter == 0) {
    int hidx[16];
#pragma unroll
    for (int r = 0; r < 16; ++r) hidx[r] = (row0 + r) >> 3;
    tdm_gather16(pemb_bf, lds_off(&b->hb[0][0]), hidx, 8, PN);
    tdm_gather16(path_emb, lds_off(&b->hof[0][0]), hidx, 16, PN);
  } else {
    tdm_linear(ps_bf + (size_t)row0 * Dn, lds_off(&b->hb[0][0]), 128);
    tdm_linear(path_state + (size_t)row0 * Dn, lds_off(&b->hof[0][0]), 256);
  }
}

// ---------------- weight pre-pack (f32 -> bf16, B-operand lane layout) ----------------
// Layout in ws: ih_p(6144) hh_p(3072) ih_q(3072) hh_q(3072) ih_l(3072) hh_l(3072) r1(512) r2(512)
__global__ void pack_weights_kernel(const float* w_ih_p, const float* w_hh_p,
                                    const float* w_ih_q, const float* w_hh_q,
                                    const float* w_ih_l, const float* w_hh_l,
                                    const float* rw1, const float* rw2, unsigned short* dst) {
  int e = blockIdx.x * blockDim.x + threadIdx.x;
  if (e >= 22528) return;
  float v;
  if (e < 21504) {
    const float* W; int Kin; int idx;
    if (e < 6144)       { W = w_ih_p; Kin = 64; idx = e; }
    else if (e < 9216)  { W = w_hh_p; Kin = 32; idx = e - 6144; }
    else if (e < 12288) { W = w_ih_q; Kin = 32; idx = e - 9216; }
    else if (e < 15360) { W = w_hh_q; Kin = 32; idx = e - 12288; }
    else if (e < 18432) { W = w_ih_l; Kin = 32; idx = e - 15360; }
    else                { W = w_hh_l; Kin = 32; idx = e - 18432; }
    int ks = idx / 3072, rem = idx % 3072;
    int nt = rem / 512, rem2 = rem % 512;
    int lane = rem2 / 16, j = rem2 % 16;
    int K = ks * 32 + ((lane >> 4) << 3) + (j & 7) + ((j >> 3) << 4);  // B layout: lane<16 K 0..7,16..23; lane>=16 K 8..15,24..31
    int N = nt * 16 + (lane & 15);
    v = W[N * Kin + K];
  } else {
    int idx = e - 21504;
    int mtx = idx / 512, rem2 = idx % 512;
    int lane = rem2 / 16, j = rem2 % 16;
    int K = ((lane >> 4) << 3) + (j & 7) + ((j >> 3) << 4);
    int N = lane & 15;
    v = (mtx == 0) ? rw1[N * 32 + K] : ((K < 16) ? rw2[N * 16 + K] : 0.0f);
  }
  dst[e] = __builtin_bit_cast(unsigned short, f2bf(v));
}

// ---------------- embedding MLPs ----------------
template <int NIN>
__device__ __forceinline__ void mlp2(const float* pf, const float* w1, const float* b1,
                                     const float* w2, const float* b2, float* outv) {
  float h1[32];
#pragma unroll
  for (int o = 0; o < 32; ++o) {
    float s = b1[o];
    for (int k = 0; k < NIN; ++k) s += w1[o * NIN + k] * pf[k];
    h1[o] = fmaxf(s, 0.0f);
  }
#pragma unroll
  for (int o = 0; o < 32; ++o) {
    float s = b2[o];
    for (int k = 0; k < 32; ++k) s += w2[o * 32 + k] * h1[k];
    outv[o] = fmaxf(s, 0.0f);
  }
}

__global__ void path_emb_kernel(const float* traffic, const float* packets,
                                const float* eq_lambda, const float* avg_pkts_lambda,
                                const float* exp_max_factor, const float* pkts_lambda_on,
                                const float* avg_t_off, const float* avg_t_on,
                                const float* ar_a, const float* sigma, const int* model,
                                const float* w1, const float* b1, const float* w2,
                                const float* b2, float* path_emb, __bf16* pemb_bf) {
  int p = blockIdx.x * blockDim.x + threadIdx.x;
  if (p >= PN) return;
  float pf[17];
  pf[0] = (traffic[p] - 1385.4058837890625f) / 859.8118896484375f;
  pf[1] = (packets[p] - 1.4015231132507324f) / 0.8932565450668335f;
  int md = model[p];
#pragma unroll
  for (int k = 0; k < 7; ++k) pf[2 + k] = (md == k) ? 1.0f : 0.0f;
  pf[9]  = (eq_lambda[p]       - 1350.97119140625f)   / 858.316162109375f;
  pf[10] = (avg_pkts_lambda[p] - 0.9117304086685181f) / 0.9723503589630127f;
  pf[11] = (exp_max_factor[p]  - 6.663637638092041f)  / 4.715115070343018f;
  pf[12] = (pkts_lambda_on[p]  - 0.9116322994232178f) / 1.651275396347046f;
  pf[13] = (avg_t_off[p]       - 1.6649284362792969f) / 2.356407403945923f;
  pf[14] = (avg_t_on[p]        - 1.6649284362792969f) / 2.356407403945923f;
  pf[15] = ar_a[p];
  pf[16] = sigma[p];
  float o[32];
  mlp2<17>(pf, w1, b1, w2, b2, o);
#pragma unroll
  for (int k = 0; k < 32; ++k) {
    path_emb[(size_t)p * Dn + k] = o[k];
    pemb_bf[(size_t)p * Dn + k] = f2bf(o[k]);
  }
}

__global__ void link_emb_kernel(const float* traffic, const float* capacity, const int* policy,
                                const int* path_to_link, const float* w1, const float* b1,
                                const float* w2, const float* b2, float* link_state,
                                __bf16* ls_bf) {
  int l = blockIdx.x * blockDim.x + threadIdx.x;
  if (l >= LN) return;
  float load = 0.0f;
  for (int k = 0; k < KLn; ++k) load += traffic[path_to_link[(size_t)l * KLn + k]];
  load /= capacity[l];
  float pf[5];
  pf[0] = load;
  int pol = policy[l];
#pragma unroll
  for (int k = 0; k < 4; ++k) pf[1 + k] = (pol == k) ? 1.0f : 0.0f;
  float o[32];
  mlp2<5>(pf, w1, b1, w2, b2, o);
#pragma unroll
  for (int k = 0; k < 32; ++k) {
    link_state[(size_t)l * Dn + k] = o[k];
    ls_bf[(size_t)l * Dn + k] = f2bf(o[k]);
  }
}

__global__ void queue_emb_kernel(const float* queue_size, const int* priority,
                                 const float* weight, const float* w1, const float* b1,
                                 const float* w2, const float* b2, float* queue_state,
                                 __bf16* qs_bf) {
  int q = blockIdx.x * blockDim.x + threadIdx.x;
  if (q >= QN) return;
  float pf[5];
  pf[0] = (queue_size[q] - 30259.10546875f) / 21410.095703125f;
  int pr = priority[q];
#pragma unroll
  for (int k = 0; k < 3; ++k) pf[1 + k] = (pr == k) ? 1.0f : 0.0f;
  pf[4] = weight[q];
  float o[32];
  mlp2<5>(pf, w1, b1, w2, b2, o);
#pragma unroll
  for (int k = 0; k < 32; ++k) {
    queue_state[(size_t)q * Dn + k] = o[k];
    qs_bf[(size_t)q * Dn + k] = f2bf(o[k]);
  }
}

// ------- path GRU + fused WMMA readout: persistent waves, TDM double-buffered tiles -------
__global__ void __launch_bounds__(32) gru_path_kernel(
    int iter, float* path_state, __bf16* ps_bf, const float* path_emb, const __bf16* pemb_bf,
    const __bf16* qs_bf, const __bf16* ls_bf, const int* q2p, const int* l2p,
    const unsigned short* pack, const float* bih, const float* bhh, const float* rb1,
    const float* rb2, const float* rw3, const float* rb3, const int* length,
    const float* capacity, float* out) {
  __shared__ TileBuf tb[2];
  __shared__ __bf16 htb[16][Dn];
  __shared__ __bf16 ro1b[16][Dn];
  __shared__ float  ro2[16][16];
  __shared__ unsigned short bsh[10240];  // ih(6144) hh(3072) r1(512) r2(512)
  int lane = threadIdx.x;

  {  // stage all B operands into LDS once per block
    const uint4* s1 = (const uint4*)pack;  // ih_p + hh_p contiguous: 9216 ushort
    uint4* d1 = (uint4*)bsh;
    for (int i = lane; i < 1152; i += 32) d1[i] = s1[i];
    const uint4* s2 = (const uint4*)(pack + 21504);  // r1 + r2: 1024 ushort
    uint4* d2 = (uint4*)(bsh + 9216);
    for (int i = lane; i < 128; i += 32) d2[i] = s2[i];
  }
  const unsigned short* sb_ih = bsh;
  const unsigned short* sb_hh = bsh + 6144;
  const unsigned short* sb_r1 = bsh + 9216;
  const unsigned short* sb_r2 = bsh + 9728;

  int c0 = lane & 15, hi = lane >> 4;
  float brA[2], bzA[2], binA[2], bhnA[2];
#pragma unroll
  for (int half = 0; half < 2; ++half) {
    int c = c0 + 16 * half;
    brA[half]  = bih[c] + bhh[c];
    bzA[half]  = bih[32 + c] + bhh[32 + c];
    binA[half] = bih[64 + c];
    bhnA[half] = bhh[64 + c];
  }
  float b1c = rb1[c0], b2c = rb2[c0], b3s = rb3[0];
  float w3v[16];
#pragma unroll
  for (int k = 0; k < 16; ++k) w3v[k] = rw3[k];

  const int NT = (PN * TN) / 16;
  int tile = blockIdx.x;
  int cur = 0;
  if (tile < NT)
    stage_path_tile(iter, tile * 16, &tb[0], path_state, ps_bf, path_emb, pemb_bf, qs_bf, ls_bf,
                    q2p, l2p);
  for (; tile < NT; tile += gridDim.x) {
    int nxt = tile + gridDim.x;
    if (nxt < NT) {  // prefetch next tile into the other buffer, overlap with compute
      stage_path_tile(iter, nxt * 16, &tb[cur ^ 1], path_state, ps_bf, path_emb, pemb_bf, qs_bf,
                      ls_bf, q2p, l2p);
      __builtin_amdgcn_s_wait_tensorcnt(4);  // current tile's 4 TDM ops complete (in-order)
    } else {
      __builtin_amdgcn_s_wait_tensorcnt(0);
    }
    __syncthreads();
    TileBuf* B = &tb[cur];
    int row0 = tile * 16;

    v8f z8 = {0.f, 0.f, 0.f, 0.f, 0.f, 0.f, 0.f, 0.f};
    v8f acc_i[6], acc_h[6];
#pragma unroll
    for (int g = 0; g < 6; ++g) { acc_i[g] = z8; acc_h[g] = z8; }
    const __bf16* xs[2] = {&B->xq[0][0], &B->xl[0][0]};
    gru_accum<2>(xs, &B->hb[0][0], sb_ih, sb_hh, lane, acc_i, acc_h);
    gru_gates<true>(acc_i, acc_h, brA, bzA, binA, bhnA, B->hof, path_state, ps_bf, htb, row0,
                    lane);
    __syncthreads();

    // readout: WMMA 32->16, WMMA 16(pad32)->16, scalar 16->1, jitter over hops
    v16bf a_h = load_a_bf(&htb[0][0], Dn, lane, 0);
    v8f r1 = wmma_bf16(a_h, load_b(sb_r1, 0, 0, lane), z8);
#pragma unroll
    for (int r = 0; r < 8; ++r) {
      int m = r + (hi << 3);
      ro1b[m][c0] = f2bf(fmaxf(r1[r] + b1c, 0.0f));
      ro1b[m][c0 + 16] = __builtin_bit_cast(__bf16, (unsigned short)0);  // zero K pad
    }
    __syncthreads();
    v16bf a_1 = load_a_bf(&ro1b[0][0], Dn, lane, 0);
    v8f r2 = wmma_bf16(a_1, load_b(sb_r2, 0, 0, lane), z8);
#pragma unroll
    for (int r = 0; r < 8; ++r) {
      int m = r + (hi << 3);
      ro2[m][c0] = fmaxf(r2[r] + b2c, 0.0f);
    }
    __syncthreads();
    if (lane < 16) {
      float s = b3s;
#pragma unroll
      for (int k = 0; k < 16; ++k) s += w3v[k] * ro2[lane][k];
      int row = row0 + lane;
      int p = row >> 3, t = row & 7;
      float contrib = 0.0f;
      if (t < length[p]) contrib = s / capacity[l2p[row]];
      contrib += __shfl_down(contrib, 4, 8);
      contrib += __shfl_down(contrib, 2, 8);
      contrib += __shfl_down(contrib, 1, 8);
      if (t == 0) out[(size_t)p * ITERS + iter] = contrib;  // exclusive ownership
    }
    __syncthreads();
    cur ^= 1;
  }
}

// ---------------- queue GRU: f32 gather-sum over KQ paths, then GRU ----------------
__global__ void __launch_bounds__(32) gru_queue_kernel(
    const float* path_state, float* queue_state, __bf16* qs_bf, const int* p2q,
    const unsigned short* pack, const float* bih, const float* bhh) {
  __shared__ __bf16 xb[16][Dn];
  __shared__ __bf16 hbt[16][Dn];
  __shared__ float  hof[16][Dn];
  __shared__ unsigned short bsh[6144];  // ih_q + hh_q
  int lane = threadIdx.x;
  {
    const uint4* s1 = (const uint4*)(pack + 9216);
    uint4* d1 = (uint4*)bsh;
    for (int i = lane; i < 768; i += 32) d1[i] = s1[i];
  }
  const unsigned short* sb_ih = bsh;
  const unsigned short* sb_hh = bsh + 3072;
  int c0 = lane & 15;
  float brA[2], bzA[2], binA[2], bhnA[2];
#pragma unroll
  for (int half = 0; half < 2; ++half) {
    int c = c0 + 16 * half;
    brA[half]  = bih[c] + bhh[c];
    bzA[half]  = bih[32 + c] + bhh[32 + c];
    binA[half] = bih[64 + c];
    bhnA[half] = bhh[64 + c];
  }
  __syncthreads();
  for (int tile = blockIdx.x; tile < QN / 16; tile += gridDim.x) {
    int row0 = tile * 16;
    for (int r = 0; r < 16; ++r) {
      int q = row0 + r;
      int myidx = p2q[(size_t)q * KQn + lane];  // lane k holds gather index k
      float acc = 0.0f;
#pragma unroll 8
      for (int k = 0; k < KQn; ++k) {
        int idx = __shfl(myidx, k, 32);
        acc += path_state[(size_t)idx * Dn + lane];
      }
      xb[r][lane] = f2bf(acc);
      hof[r][lane] = queue_state[(size_t)q * Dn + lane];
      hbt[r][lane] = qs_bf[(size_t)q * Dn + lane];
    }
    __syncthreads();
    v8f z8 = {0.f, 0.f, 0.f, 0.f, 0.f, 0.f, 0.f, 0.f};
    v8f acc_i[6], acc_h[6];
#pragma unroll
    for (int g = 0; g < 6; ++g) { acc_i[g] = z8; acc_h[g] = z8; }
    const __bf16* xs[1] = {&xb[0][0]};
    gru_accum<1>(xs, &hbt[0][0], sb_ih, sb_hh, lane, acc_i, acc_h);
    gru_gates<false>(acc_i, acc_h, brA, bzA, binA, bhnA, hof, queue_state, qs_bf, nullptr, row0,
                     lane);
    __syncthreads();
  }
}

// ---------------- link GRU ----------------
__global__ void __launch_bounds__(32) gru_link_kernel(
    const __bf16* qs_bf, float* link_state, __bf16* ls_bf, const int* q2l,
    const unsigned short* pack, const float* bih, const float* bhh) {
  __shared__ __bf16 xb[16][Dn];
  __shared__ __bf16 hbt[16][Dn];
  __shared__ float  hof[16][Dn];
  __shared__ unsigned short bsh[6144];  // ih_l + hh_l
  int lane = threadIdx.x;
  {
    const uint4* s1 = (const uint4*)(pack + 15360);
    uint4* d1 = (uint4*)bsh;
    for (int i = lane; i < 768; i += 32) d1[i] = s1[i];
  }
  const unsigned short* sb_ih = bsh;
  const unsigned short* sb_hh = bsh + 3072;
  int c0 = lane & 15;
  float brA[2], bzA[2], binA[2], bhnA[2];
#pragma unroll
  for (int half = 0; half < 2; ++half) {
    int c = c0 + 16 * half;
    brA[half]  = bih[c] + bhh[c];
    bzA[half]  = bih[32 + c] + bhh[32 + c];
    binA[half] = bih[64 + c];
    bhnA[half] = bhh[64 + c];
  }
  __syncthreads();
  for (int tile = blockIdx.x; tile < LN / 16; tile += gridDim.x) {
    int row0 = tile * 16;
#pragma unroll 4
    for (int r = 0; r < 16; ++r) {
      int row = row0 + r;
      int q = q2l[row];
      xb[r][lane] = qs_bf[(size_t)q * Dn + lane];
      hof[r][lane] = link_state[(size_t)row * Dn + lane];
      hbt[r][lane] = ls_bf[(size_t)row * Dn + lane];
    }
    __syncthreads();
    v8f z8 = {0.f, 0.f, 0.f, 0.f, 0.f, 0.f, 0.f, 0.f};
    v8f acc_i[6], acc_h[6];
#pragma unroll
    for (int g = 0; g < 6; ++g) { acc_i[g] = z8; acc_h[g] = z8; }
    const __bf16* xs[1] = {&xb[0][0]};
    gru_accum<1>(xs, &hbt[0][0], sb_ih, sb_hh, lane, acc_i, acc_h);
    gru_gates<false>(acc_i, acc_h, brA, bzA, binA, bhnA, hof, link_state, ls_bf, nullptr, row0,
                     lane);
    __syncthreads();
  }
}

extern "C" void kernel_launch(void* const* d_in, const int* in_sizes, int n_in, void* d_out,
                              int out_size, void* d_ws, size_t ws_size, hipStream_t stream) {
  (void)in_sizes; (void)n_in; (void)out_size; (void)ws_size;
  const float* traffic         = (const float*)d_in[0];
  const float* packets         = (const float*)d_in[1];
  const float* eq_lambda       = (const float*)d_in[2];
  const float* avg_pkts_lambda = (const float*)d_in[3];
  const float* exp_max_factor  = (const float*)d_in[4];
  const float* pkts_lambda_on  = (const float*)d_in[5];
  const float* avg_t_off       = (const float*)d_in[6];
  const float* avg_t_on        = (const float*)d_in[7];
  const float* ar_a            = (const float*)d_in[8];
  const float* sigma           = (const float*)d_in[9];
  const float* capacity        = (const float*)d_in[10];
  const float* queue_size      = (const float*)d_in[11];
  const float* weight          = (const float*)d_in[12];
  const int* length   = (const int*)d_in[13];
  const int* model    = (const int*)d_in[14];
  const int* policy   = (const int*)d_in[15];
  const int* priority = (const int*)d_in[16];
  const int* q2p = (const int*)d_in[17];
  const int* l2p = (const int*)d_in[18];
  const int* p2l = (const int*)d_in[19];
  const int* p2q = (const int*)d_in[20];
  const int* q2l = (const int*)d_in[21];
  const float* pe_w1 = (const float*)d_in[22];
  const float* pe_b1 = (const float*)d_in[23];
  const float* pe_w2 = (const float*)d_in[24];
  const float* pe_b2 = (const float*)d_in[25];
  const float* le_w1 = (const float*)d_in[26];
  const float* le_b1 = (const float*)d_in[27];
  const float* le_w2 = (const float*)d_in[28];
  const float* le_b2 = (const float*)d_in[29];
  const float* qe_w1 = (const float*)d_in[30];
  const float* qe_b1 = (const float*)d_in[31];
  const float* qe_w2 = (const float*)d_in[32];
  const float* qe_b2 = (const float*)d_in[33];
  const float* gp_wih = (const float*)d_in[34];
  const float* gp_whh = (const float*)d_in[35];
  const float* gp_bih = (const float*)d_in[36];
  const float* gp_bhh = (const float*)d_in[37];
  const float* gq_wih = (const float*)d_in[38];
  const float* gq_whh = (const float*)d_in[39];
  const float* gq_bih = (const float*)d_in[40];
  const float* gq_bhh = (const float*)d_in[41];
  const float* gl_wih = (const float*)d_in[42];
  const float* gl_whh = (const float*)d_in[43];
  const float* gl_bih = (const float*)d_in[44];
  const float* gl_bhh = (const float*)d_in[45];
  const float* r_w1 = (const float*)d_in[46];
  const float* r_b1 = (const float*)d_in[47];
  const float* r_w2 = (const float*)d_in[48];
  const float* r_b2 = (const float*)d_in[49];
  const float* r_w3 = (const float*)d_in[50];
  const float* r_b3 = (const float*)d_in[51];

  char* ws = (char*)d_ws;
  float* path_state  = (float*)(ws);                        // 51,200,000 B
  float* queue_state = (float*)(ws + 51200000);             //  2,560,000 B
  float* link_state  = (float*)(ws + 53760000);             //  1,024,000 B
  float* path_emb    = (float*)(ws + 54784000);             //  6,400,000 B
  unsigned short* pack = (unsigned short*)(ws + 61184000);  //     45,056 B
  __bf16* ps_bf   = (__bf16*)(ws + 61229056);               // 25,600,000 B
  __bf16* qs_bf   = (__bf16*)(ws + 86829056);               //  1,280,000 B
  __bf16* ls_bf   = (__bf16*)(ws + 88109056);               //    512,000 B
  __bf16* pemb_bf = (__bf16*)(ws + 88621056);               //  3,200,000 B

  pack_weights_kernel<<<(22528 + 255) / 256, 256, 0, stream>>>(gp_wih, gp_whh, gq_wih, gq_whh,
                                                               gl_wih, gl_whh, r_w1, r_w2, pack);
  path_emb_kernel<<<(PN + 127) / 128, 128, 0, stream>>>(
      traffic, packets, eq_lambda, avg_pkts_lambda, exp_max_factor, pkts_lambda_on, avg_t_off,
      avg_t_on, ar_a, sigma, model, pe_w1, pe_b1, pe_w2, pe_b2, path_emb, pemb_bf);
  link_emb_kernel<<<(LN + 127) / 128, 128, 0, stream>>>(traffic, capacity, policy, p2l, le_w1,
                                                        le_b1, le_w2, le_b2, link_state, ls_bf);
  queue_emb_kernel<<<(QN + 127) / 128, 128, 0, stream>>>(queue_size, priority, weight, qe_w1,
                                                         qe_b1, qe_w2, qe_b2, queue_state, qs_bf);
  for (int it = 0; it < ITERS; ++it) {
    gru_path_kernel<<<6250, 32, 0, stream>>>(it, path_state, ps_bf, path_emb, pemb_bf, qs_bf,
                                             ls_bf, q2p, l2p, pack, gp_bih, gp_bhh, r_b1, r_b2,
                                             r_w3, r_b3, length, capacity, (float*)d_out);
    gru_queue_kernel<<<625, 32, 0, stream>>>(path_state, queue_state, qs_bf, p2q, pack, gq_bih,
                                             gq_bhh);
    gru_link_kernel<<<250, 32, 0, stream>>>(qs_bf, link_state, ls_bf, q2l, pack, gl_bih,
                                            gl_bhh);
  }
}
